// WideGRU_52982716563919
// MI455X (gfx1250) — compile-verified, matrix operand order checked by reading, not run on hardware
//
#include <hip/hip_runtime.h>
#include <hip/hip_bf16.h>
#include <math.h>

// Problem constants
#define NN 16      // number of GRU cells
#define II 128     // input size per cell
#define HH 128     // hidden size per cell
#define BB 16      // batch
#define TT 512     // time steps
#define G3 384     // 3*H
#define MTOT (BB*TT)   // 8192 rows for input projection

typedef __attribute__((ext_vector_type(16))) _Float16 v16h;
typedef __attribute__((ext_vector_type(8)))  _Float16 v8h;
typedef __attribute__((ext_vector_type(8)))  float    v8f;

#if __has_builtin(__builtin_amdgcn_global_load_async_to_lds_b128) && \
    __has_builtin(__builtin_amdgcn_s_wait_asynccnt)
#define HAVE_ASYNC 1
#else
#define HAVE_ASYNC 0
#endif

#if HAVE_ASYNC
typedef int v4i_t __attribute__((ext_vector_type(4)));
typedef v4i_t __attribute__((address_space(1))) g_v4i;   // global-AS 4xint
typedef v4i_t __attribute__((address_space(3))) l_v4i;   // LDS-AS 4xint
#endif

__device__ __forceinline__ float sigmoid_f(float x) {
    return 1.0f / (1.0f + __expf(-x));
}

// ---------------------------------------------------------------------------
// f32 -> f16 conversion (8 elements / thread), for x, W_ih, W_hh
// ---------------------------------------------------------------------------
__global__ __launch_bounds__(256) void f32_to_f16_k(const float* __restrict__ s,
                                                    _Float16* __restrict__ d,
                                                    int n8) {
    int i = blockIdx.x * blockDim.x + threadIdx.x;
    if (i < n8) {
        const float4* sp = (const float4*)s + (size_t)i * 2;
        float4 a = sp[0];
        float4 b = sp[1];
        v8h o;
        o[0] = (_Float16)a.x; o[1] = (_Float16)a.y;
        o[2] = (_Float16)a.z; o[3] = (_Float16)a.w;
        o[4] = (_Float16)b.x; o[5] = (_Float16)b.y;
        o[6] = (_Float16)b.z; o[7] = (_Float16)b.w;
        ((v8h*)d)[i] = o;
    }
}

// ---------------------------------------------------------------------------
// Phase 1: gates_x[n][t][b][g] = x[b,t,n*I: ] . W_ih[n,g,:] + bias_ih[n,g]
// Grid: (64 m-blocks of 128 rows, 16 cells), 256 threads = 8 waves.
// ---------------------------------------------------------------------------
__global__ __launch_bounds__(256) void gru_xproj_kernel(
    const _Float16* __restrict__ xh,        // [8192][2048] f16
    const _Float16* __restrict__ wih,       // [N][384][128] f16
    const float*    __restrict__ bias_ih,   // [N][384]
    float*          __restrict__ gates_x)   // [N][T][B][384]
{
    const int n    = blockIdx.y;
    const int lane = threadIdx.x & 31;
    const int wave = threadIdx.x >> 5;
    const int lo   = lane & 15;
    const int k0   = (lane < 16) ? 0 : 8;    // per-lane K chunk base (WMMA layout)
    const int mhi  = (lane < 16) ? 0 : 8;    // D row offset for hi lanes
    const int m_base = blockIdx.x * 128 + wave * 16;

    // A tiles: lane holds row (m_base+lo), K chunks {k0..k0+7, k0+16..k0+23}
    v16h A[4];
    const _Float16* xp = xh + (size_t)(m_base + lo) * (NN * II) + n * II + k0;
#pragma unroll
    for (int kk = 0; kk < 4; ++kk) {
        const _Float16* p = xp + kk * 32;
        v16h a;
#pragma unroll
        for (int i = 0; i < 8; ++i) { a[i] = p[i]; a[8 + i] = p[16 + i]; }
        A[kk] = a;
    }

    for (int gt = 0; gt < 24; ++gt) {
        float bias = bias_ih[n * G3 + gt * 16 + lo];
        v8f acc;
#pragma unroll
        for (int j = 0; j < 8; ++j) acc[j] = bias;

        const _Float16* wp = wih + ((size_t)n * G3 + gt * 16 + lo) * II + k0;
#pragma unroll
        for (int kk = 0; kk < 4; ++kk) {
            const _Float16* p = wp + kk * 32;
            v16h b;
#pragma unroll
            for (int i = 0; i < 8; ++i) { b[i] = p[i]; b[8 + i] = p[16 + i]; }
            acc = __builtin_amdgcn_wmma_f32_16x16x32_f16(
                false, A[kk], false, b, (short)0, acc, false, false);
        }

#pragma unroll
        for (int j = 0; j < 8; ++j) {
            int m = m_base + j + mhi;
            int b = m >> 9;        // batch (T = 512)
            int t = m & 511;
            gates_x[(((size_t)n * TT + t) * BB + b) * G3 + gt * 16 + lo] = acc[j];
        }
    }
}

// ---------------------------------------------------------------------------
// Phase 2: recurrent scan. One block per cell n, 8 waves.
// Wave w owns hidden units [16w,16w+16): its 3 g-tiles are the r/z/n tiles
// for those units -> gate combine is wave-local. W_hh pinned in registers.
// gates_x slices double-buffered into LDS via async-to-LDS copies.
// ---------------------------------------------------------------------------
#define HSTRIDE 132   // padded LDS row stride (floats) to break bank conflicts

#if HAVE_ASYNC
// Each wave copies its 1/8 slice (6 x b128 per lane) of one 24.5KB gx step.
__device__ __forceinline__ void async_copy_step(const float* __restrict__ src,
                                                float* dst_lds,
                                                int wave, int lane) {
#pragma unroll
    for (int i = 0; i < 6; ++i) {
        int dw = (wave * 192 + i * 32 + lane) * 4;   // dword offset
        __builtin_amdgcn_global_load_async_to_lds_b128(
            (g_v4i*)(src + dw),
            (l_v4i*)(dst_lds + dw), 0, 0);
    }
}
#endif

__global__ __launch_bounds__(256) void gru_scan_kernel(
    const float*    __restrict__ gates_x,   // [N][T][B][384]
    const _Float16* __restrict__ whh,       // [N][384][128] f16
    const float*    __restrict__ bias_hh,   // [N][384]
    const float*    __restrict__ h0,        // [B][N*H]
    float*          __restrict__ out,       // [B][T][N*H]
    float*          __restrict__ hn)        // [B][N*H]
{
    const int n    = blockIdx.x;
    const int lane = threadIdx.x & 31;
    const int wave = threadIdx.x >> 5;
    const int lo   = lane & 15;
    const int k0   = (lane < 16) ? 0 : 8;
    const int bhi  = (lane < 16) ? 0 : 8;
    const int unit = wave * 16 + lo;

    __shared__ float h_lds[BB * HSTRIDE];
#if HAVE_ASYNC
    __shared__ __align__(16) float gxbuf[2][BB * G3];   // 2 x 24.5KB ping-pong
#endif

    // Pin W_hh B-tiles in registers: Bt[s][kk]
    v16h Bt[3][4];
#pragma unroll
    for (int s = 0; s < 3; ++s) {
        const _Float16* wp = whh + ((size_t)n * G3 + s * HH + unit) * HH + k0;
#pragma unroll
        for (int kk = 0; kk < 4; ++kk) {
            const _Float16* p = wp + kk * 32;
            v16h b;
#pragma unroll
            for (int i = 0; i < 8; ++i) { b[i] = p[i]; b[8 + i] = p[16 + i]; }
            Bt[s][kk] = b;
        }
    }

    float bias[3];
#pragma unroll
    for (int s = 0; s < 3; ++s) bias[s] = bias_hh[n * G3 + s * HH + unit];

    const float* gx_base = gates_x + (size_t)n * TT * BB * G3;

#if HAVE_ASYNC
    async_copy_step(gx_base, &gxbuf[0][0], wave, lane);   // prefetch t = 0
#endif

    // Init h: register slice (D-tile layout) + LDS copy
    v8f hreg;
#pragma unroll
    for (int j = 0; j < 8; ++j) {
        int b = j + bhi;
        float v = h0[(size_t)b * (NN * HH) + n * HH + unit];
        hreg[j] = v;
        h_lds[b * HSTRIDE + unit] = v;
    }
    __syncthreads();

    for (int t = 0; t < TT; ++t) {
        // Build A tiles (batch x H) from LDS h, f32 -> f16
        v16h A[4];
#pragma unroll
        for (int kk = 0; kk < 4; ++kk) {
            const float* p = &h_lds[lo * HSTRIDE + kk * 32 + k0];
            v16h a;
#pragma unroll
            for (int i = 0; i < 8; ++i) {
                a[i]     = (_Float16)p[i];
                a[8 + i] = (_Float16)p[16 + i];
            }
            A[kk] = a;
        }

#if HAVE_ASYNC
        // Kick off next step's gx copy, then ensure this step's slice landed:
        // async ops complete in order, so asynccnt<=6 leaves only the 6 newest
        // (t+1's) outstanding.
        if (t + 1 < TT) {
            async_copy_step(gx_base + (size_t)(t + 1) * BB * G3,
                            &gxbuf[(t + 1) & 1][0], wave, lane);
            __builtin_amdgcn_s_wait_asynccnt(6);
        } else {
            __builtin_amdgcn_s_wait_asynccnt(0);
        }
#endif
        __syncthreads();   // h reads done; gx[t] visible workgroup-wide

#if HAVE_ASYNC
        const float* gx_t = &gxbuf[t & 1][0];
#else
        const float* gx_t = gx_base + (size_t)t * BB * G3;
        if (t + 1 < TT)
            __builtin_prefetch(gx_t + BB * G3 + (size_t)(lane * 384), 0, 1);
#endif

        // gates_h = h . W_hh^T + bias_hh  (C init = bias)
        v8f acc[3];
#pragma unroll
        for (int s = 0; s < 3; ++s) {
            v8f c;
#pragma unroll
            for (int j = 0; j < 8; ++j) c[j] = bias[s];
#pragma unroll
            for (int kk = 0; kk < 4; ++kk) {
                c = __builtin_amdgcn_wmma_f32_16x16x32_f16(
                    false, A[kk], false, Bt[s][kk], (short)0, c, false, false);
            }
            acc[s] = c;
        }

        // Gate math + state update (f32)
        v8f hnew;
#pragma unroll
        for (int j = 0; j < 8; ++j) {
            int b = j + bhi;
            const float* g = gx_t + b * G3;
            float gxr = g[unit];
            float gxz = g[HH + unit];
            float gxn = g[2 * HH + unit];
            float r  = sigmoid_f(gxr + acc[0][j]);
            float z  = sigmoid_f(gxz + acc[1][j]);
            float nv = tanhf(gxn + r * acc[2][j]);
            float h_ = (1.0f - z) * nv + z * hreg[j];
            hnew[j] = h_;
            out[((size_t)b * TT + t) * (NN * HH) + n * HH + unit] = h_;
            h_lds[b * HSTRIDE + unit] = h_;
        }
        hreg = hnew;
        __syncthreads();   // h (and gx-buffer reads) complete before next step
    }

    // Final hidden state
#pragma unroll
    for (int j = 0; j < 8; ++j) {
        int b = j + bhi;
        hn[(size_t)b * (NN * HH) + n * HH + unit] = hreg[j];
    }
}

// ---------------------------------------------------------------------------
// Launcher
// ---------------------------------------------------------------------------
extern "C" void kernel_launch(void* const* d_in, const int* in_sizes, int n_in,
                              void* d_out, int out_size, void* d_ws, size_t ws_size,
                              hipStream_t stream) {
    const float* x   = (const float*)d_in[0];   // [B, T, N*I]
    const float* h0  = (const float*)d_in[1];   // [1, B, N*H]
    const float* wih = (const float*)d_in[2];   // [N, 3H, I]
    const float* whh = (const float*)d_in[3];   // [N, 3H, H]
    const float* bih = (const float*)d_in[4];   // [N, 3H]
    const float* bhh = (const float*)d_in[5];   // [N, 3H]

    float* out = (float*)d_out;                         // [B, T, N*H]
    float* hn  = out + (size_t)BB * TT * (NN * HH);     // [1, B, N*H]

    // Workspace layout
    char* ws = (char*)d_ws;
    float*    gates_x = (float*)ws;
    size_t off = (size_t)NN * TT * BB * G3 * sizeof(float);        // 201 MB
    _Float16* xh   = (_Float16*)(ws + off);
    off += (size_t)MTOT * (NN * II) * sizeof(_Float16);            // 33.5 MB
    _Float16* wihh = (_Float16*)(ws + off);
    off += (size_t)NN * G3 * II * sizeof(_Float16);                // 1.5 MB
    _Float16* whhh = (_Float16*)(ws + off);

    const int nx = MTOT * NN * II;      // 16,777,216
    const int nw = NN * G3 * II;        // 786,432

    f32_to_f16_k<<<(nx / 8 + 255) / 256, 256, 0, stream>>>(x,   xh,   nx / 8);
    f32_to_f16_k<<<(nw / 8 + 255) / 256, 256, 0, stream>>>(wih, wihh, nw / 8);
    f32_to_f16_k<<<(nw / 8 + 255) / 256, 256, 0, stream>>>(whh, whhh, nw / 8);

    gru_xproj_kernel<<<dim3(MTOT / 128, NN), 256, 0, stream>>>(xh, wihh, bih, gates_x);

    gru_scan_kernel<<<NN, 256, 0, stream>>>(gates_x, whhh, bhh, h0, out, hn);
}
